// CrossAttention_76261439308078
// MI455X (gfx1250) — matrix operand order repaired and checked by reading, not compiled, not run on hardware
//
#include <hip/hip_runtime.h>

typedef __attribute__((ext_vector_type(16))) _Float16 v16h;
typedef __attribute__((ext_vector_type(8)))  _Float16 v8h_t;
typedef __attribute__((ext_vector_type(2)))  _Float16 h2;
typedef __attribute__((ext_vector_type(8)))  float    v8f;
typedef __attribute__((ext_vector_type(4)))  int      v4i;

#define Bn   4
#define Qn   1024
#define Kn   2048
#define Dn   256
#define Hn   8
#define HDn  32
#define CPBH 512

#if __has_builtin(__builtin_elementwise_maxnum)
#define H2MAX(a, b) __builtin_elementwise_maxnum((a), (b))
#else
#define H2MAX(a, b) __builtin_elementwise_max((a), (b))
#endif

__device__ __forceinline__ v8f wmma_f16(v16h a, v16h b, v8f c) {
    return __builtin_amdgcn_wmma_f32_16x16x32_f16(
        /*neg_a=*/false, a, /*neg_b=*/false, b,
        /*c_mod=*/(short)0, c, /*reuse_a=*/false, /*reuse_b=*/false);
}

// A fragment (16x32 f16, row-major source): lane<16 -> row M=lane, K {0..15};
// lane>=16 -> same row's K {16..31}.
__device__ __forceinline__ v16h load_a_frag(const _Float16* __restrict__ A,
                                            int lda, int row0, int k0) {
    int lane = threadIdx.x & 31;
    int m = lane & 15, sel = lane >> 4;
    const _Float16* p = A + (size_t)(row0 + m) * lda + k0 + sel * 16;
    union { v16h v; v8h_t h[2]; } u;
    u.h[0] = *(const v8h_t*)p;
    u.h[1] = *(const v8h_t*)(p + 8);
    return u.v;
}

// B fragment (32x16): W stored row-major [N, Kdim]; lane holds column N=lane&15,
// 16 consecutive K starting at k0 + (lane>>4)*16  -> one contiguous 32B load.
__device__ __forceinline__ v16h load_b_frag(const _Float16* __restrict__ W,
                                            int ldw, int col0, int k0) {
    int lane = threadIdx.x & 31;
    int n = lane & 15, sel = lane >> 4;
    return *(const v16h*)(W + (size_t)(col0 + n) * ldw + k0 + sel * 16);
}

// ---------------- conversion / packing ----------------

__global__ void cvt_f16_kernel(const float* __restrict__ in,
                               _Float16* __restrict__ out, int n) {
    int i = blockIdx.x * 256 + threadIdx.x;
    if (i < n) out[i] = (_Float16)in[i];
}

// cpb_w2 [H=8, 512] -> WMMA-B-swizzled [chunk c(16)][lane(32)][16 halves],
// columns 8..15 zero-padded.
__global__ void pack_w2_kernel(const float* __restrict__ w2,
                               _Float16* __restrict__ out) {
    int idx = blockIdx.x * 256 + threadIdx.x;
    if (idx >= 16 * 32 * 16) return;
    int c = idx >> 9;
    int l = (idx >> 4) & 31;
    int i = idx & 15;
    int n = l & 15;
    int k = c * 32 + (l >> 4) * 16 + i;
    out[idx] = (n < Hn) ? (_Float16)w2[n * CPBH + k] : (_Float16)0.0f;
}

// ---------------- projection GEMMs ----------------

__device__ __forceinline__ v8f gemm_tile(const _Float16* A, int lda,
                                         const _Float16* W, int ldw,
                                         int row0, int col0, int kdim) {
    v8f acc = {};
    for (int k0 = 0; k0 < kdim; k0 += 32)
        acc = wmma_f16(load_a_frag(A, lda, row0, k0),
                       load_b_frag(W, ldw, col0, k0), acc);
    return acc;
}

__global__ __launch_bounds__(256) void qproj_kernel(
    const _Float16* __restrict__ x, const _Float16* __restrict__ w,
    const float* __restrict__ bias, _Float16* __restrict__ out) {
    int wave = threadIdx.x >> 5, lane = threadIdx.x & 31;
    int t = blockIdx.x * 8 + wave;            // 4096 tiles: 256 row x 16 col
    int row0 = (t >> 4) * 16, col0 = (t & 15) * 16;
    v8f acc = gemm_tile(x, Dn, w, Dn, row0, col0, Dn);
    int n = lane & 15, sel = lane >> 4;
    float bb = bias[col0 + n];
    #pragma unroll
    for (int r = 0; r < 8; ++r)
        out[(size_t)(row0 + r + 8 * sel) * Dn + col0 + n] = (_Float16)(acc[r] + bb);
}

__global__ __launch_bounds__(256) void kvproj_kernel(
    const _Float16* __restrict__ x, const _Float16* __restrict__ w,
    const float* __restrict__ bias, _Float16* __restrict__ kout,
    _Float16* __restrict__ vT) {
    int wave = threadIdx.x >> 5, lane = threadIdx.x & 31;
    int t = blockIdx.x * 8 + wave;            // 16384 tiles: 512 row x 32 col
    int row0 = (t >> 5) * 16, col0 = (t & 31) * 16;
    v8f acc = gemm_tile(x, Dn, w, Dn, row0, col0, Dn);
    int n = lane & 15, sel = lane >> 4;
    int col = col0 + n;
    float bb = bias[col];
    #pragma unroll
    for (int r = 0; r < 8; ++r) {
        int row = row0 + r + 8 * sel;          // b*2048 + key
        float v = acc[r] + bb;
        if (col < Dn) {
            kout[(size_t)row * Dn + col] = (_Float16)v;
        } else {
            int b = row >> 11, key = row & (Kn - 1);
            vT[((size_t)(b * Dn + (col - Dn))) * Kn + key] = (_Float16)v;
        }
    }
}

__global__ __launch_bounds__(256) void oproj_kernel(
    const _Float16* __restrict__ x, const _Float16* __restrict__ w,
    const float* __restrict__ bias, float* __restrict__ out) {
    int wave = threadIdx.x >> 5, lane = threadIdx.x & 31;
    int t = blockIdx.x * 8 + wave;
    int row0 = (t >> 4) * 16, col0 = (t & 15) * 16;
    v8f acc = gemm_tile(x, Dn, w, Dn, row0, col0, Dn);
    int n = lane & 15, sel = lane >> 4;
    float bb = bias[col0 + n];
    #pragma unroll
    for (int r = 0; r < 8; ++r)
        out[(size_t)(row0 + r + 8 * sel) * Dn + col0 + n] = acc[r] + bb;
}

// ---------------- fused CPB-bias + flash attention ----------------
// Block: 256 threads = 8 waves, wave w == head w. Tile: 16 queries x all K.

__global__ __launch_bounds__(256) void attn_kernel(
    const _Float16* __restrict__ qproj,   // [B*Q, 256] f16
    const _Float16* __restrict__ kmat,    // [B*K, 256] f16
    const _Float16* __restrict__ vT,      // [B, 256, K] f16
    const float*    __restrict__ rd,      // [B, Q, K, 2] f32
    const float*    __restrict__ w1,      // [512, 2] f32
    const _Float16* __restrict__ w2pk,    // [16][32][16] f16 (B-swizzled)
    const unsigned char* __restrict__ msk,// [B, K] bool
    _Float16* __restrict__ attnout) {     // [B*Q, 256] f16
    __shared__ h2       w1p0s[CPBH / 2];         // packed col-0 pairs
    __shared__ h2       w1p1s[CPBH / 2];         // packed col-1 pairs
    __shared__ _Float16 w2s[16 * 32 * 16];
    __shared__ float    bias_s[512 * Hn];        // 16q x 32k pairs x 8 heads
    __shared__ _Float16 pstage[Hn][16 * 32];     // per-wave P transpose staging

    const int tid  = threadIdx.x;
    const int wave = tid >> 5;                   // head index
    const int lane = tid & 31;
    const int n    = lane & 15;
    const int sel  = lane >> 4;
    const int b    = blockIdx.y;
    const int q0   = blockIdx.x * 16;

    // pack cpb_w1 into LDS as half2 pairs (layer-1 runs on v_pk_* f16 VALU)
    for (int i = tid; i < CPBH / 2; i += 256) {
        h2 c0 = { (_Float16)w1[(2 * i) * 2],     (_Float16)w1[(2 * i + 1) * 2] };
        h2 c1 = { (_Float16)w1[(2 * i) * 2 + 1], (_Float16)w1[(2 * i + 1) * 2 + 1] };
        w1p0s[i] = c0;
        w1p1s[i] = c1;
    }
    // w2 fragments -> LDS via gfx1250 async-to-LDS path when available.
    // Probe-discovered signature: (v4i AS(1)*, v4i AS(3)*, imm offset, imm cpol).
#if __has_builtin(__builtin_amdgcn_global_load_async_to_lds_b128)
    {
        typedef __attribute__((address_space(1))) v4i gv4i_t;
        typedef __attribute__((address_space(3))) v4i lv4i_t;
        for (int i = tid; i < 1024; i += 256)    // 16 KB as 1024 x b128
            __builtin_amdgcn_global_load_async_to_lds_b128(
                (gv4i_t*)(w2pk + i * 8), (lv4i_t*)(&w2s[i * 8]), 0, 0);
        asm volatile("s_wait_asynccnt 0x0" ::: "memory");
    }
#else
    for (int i = tid; i < 16 * 32 * 16; i += 256) w2s[i] = w2pk[i];
#endif
    __syncthreads();

    // Q fragment for this head (A layout), loaded once.
    const v16h qa = load_a_frag(qproj, Dn, b * Qn + q0, wave * HDn);

    v8f accA = {}, accB = {};
    float mrow[8], lrow[8];
    #pragma unroll
    for (int r = 0; r < 8; ++r) { mrow[r] = -3.0e38f; lrow[r] = 0.0f; }

    const float scale = 0.17677669529663687f;    // 1/sqrt(32)
    const float log2e = 1.4426950408889634f;
    const v8f zf = {};
    const h2 hzero = { (_Float16)0.0f, (_Float16)0.0f };

    for (int kt = 0; kt < Kn / 32; ++kt) {
        // ---- CPB bias for 16q x 32k tile: 512 pairs split 64/wave ----
        #pragma unroll
        for (int i = 0; i < 4; ++i) {
            int rt = wave * 4 + i;               // row-tile of 16 pairs
            int pr = rt * 16 + n;                // pair id, = qq*32 + kk
            int qq = pr >> 5, kk = pr & 31;
            const float* rp =
                rd + ((size_t)(b * Qn + q0 + qq) * Kn + kt * 32 + kk) * 2;
            _Float16 rd0 = (_Float16)rp[0], rd1 = (_Float16)rp[1];
            h2 r0 = { rd0, rd0 }, r1 = { rd1, rd1 };
            v8f acc = {};
            for (int c = 0; c < 16; ++c) {       // CPB_H in chunks of 32
                union { v16h v; h2 p[8]; } ua;
                int t0 = c * 16 + sel * 8;       // (c*32 + sel*16) / 2
                #pragma unroll
                for (int t = 0; t < 4; ++t) {
                    h2 a  = r0 * w1p0s[t0 + t]     + r1 * w1p1s[t0 + t];
                    h2 bp = r0 * w1p0s[t0 + 4 + t] + r1 * w1p1s[t0 + 4 + t];
                    ua.p[t]     = H2MAX(a,  hzero);
                    ua.p[4 + t] = H2MAX(bp, hzero);
                }
                v16h bb = *(const v16h*)&w2s[(c * 32 + lane) * 16];
                acc = wmma_f16(ua.v, bb, acc);
            }
            if (n < Hn) {
                #pragma unroll
                for (int r = 0; r < 8; ++r)
                    bias_s[(rt * 16 + r + 8 * sel) * Hn + n] = acc[r];
            }
        }
        __syncthreads();

        // ---- scores for this head: Q[16,32] x K^T[32,32] ----
        const _Float16* kb =
            kmat + (size_t)(b * Kn + kt * 32) * Dn + wave * HDn + sel * 16;
        v16h kf0 = *(const v16h*)(kb + (size_t)n * Dn);
        v16h kf1 = *(const v16h*)(kb + (size_t)(n + 16) * Dn);
        v8f s0 = wmma_f16(qa, kf0, zf);
        v8f s1 = wmma_f16(qa, kf1, zf);

        float mk0 = msk[b * Kn + kt * 32 + n]      ? -3.0e38f : 0.0f;
        float mk1 = msk[b * Kn + kt * 32 + 16 + n] ? -3.0e38f : 0.0f;

        float mnew[8];
        #pragma unroll
        for (int r = 0; r < 8; ++r) {
            int qq = r + 8 * sel;
            s0[r] = s0[r] * scale + bias_s[(qq * 32 + n) * Hn + wave] + mk0;
            s1[r] = s1[r] * scale + bias_s[(qq * 32 + 16 + n) * Hn + wave] + mk1;
            float mx = fmaxf(s0[r], s1[r]);
            for (int d = 1; d < 16; d <<= 1) mx = fmaxf(mx, __shfl_xor(mx, d, 32));
            mnew[r] = fmaxf(mrow[r], mx);
        }

        #pragma unroll
        for (int r = 0; r < 8; ++r) {
            float p0 = exp2f((s0[r] - mnew[r]) * log2e);
            float p1 = exp2f((s1[r] - mnew[r]) * log2e);
            float rs = p0 + p1;
            for (int d = 1; d < 16; d <<= 1) rs += __shfl_xor(rs, d, 32);
            float corr = exp2f((mrow[r] - mnew[r]) * log2e);
            lrow[r] = lrow[r] * corr + rs;
            mrow[r] = mnew[r];
            accA[r] *= corr;
            accB[r] *= corr;
            int qq = r + 8 * sel;
            pstage[wave][qq * 32 + n]      = (_Float16)p0;
            pstage[wave][qq * 32 + 16 + n] = (_Float16)p1;
        }
        // C->A layout transpose through wave-private LDS (in-order within wave)
        v16h pa;
        {
            union { v16h v; v8h_t h[2]; } u;
            const _Float16* pp = &pstage[wave][n * 32 + sel * 16];
            u.h[0] = *(const v8h_t*)pp;
            u.h[1] = *(const v8h_t*)(pp + 8);
            pa = u.v;
        }

        // ---- PV: P[16,32] x V[32,32], V stored transposed for contiguity ----
        const _Float16* vb =
            vT + ((size_t)b * Dn + wave * HDn) * Kn + kt * 32 + sel * 16;
        v16h vf0 = *(const v16h*)(vb + (size_t)n * Kn);
        v16h vf1 = *(const v16h*)(vb + (size_t)(16 + n) * Kn);
        accA = wmma_f16(pa, vf0, accA);
        accB = wmma_f16(pa, vf1, accB);
        __syncthreads();   // bias_s reused next iteration
    }

    #pragma unroll
    for (int r = 0; r < 8; ++r) {
        float inv = 1.0f / lrow[r];
        size_t row = (size_t)(b * Qn + q0 + r + 8 * sel);
        attnout[row * Dn + wave * HDn + n]      = (_Float16)(accA[r] * inv);
        attnout[row * Dn + wave * HDn + 16 + n] = (_Float16)(accB[r] * inv);
    }
}

// ---------------- host launcher ----------------

extern "C" void kernel_launch(void* const* d_in, const int* in_sizes, int n_in,
                              void* d_out, int out_size, void* d_ws, size_t ws_size,
                              hipStream_t stream) {
    (void)in_sizes; (void)n_in; (void)out_size; (void)ws_size;
    const float* nodes  = (const float*)d_in[0];
    const float* images = (const float*)d_in[1];
    const unsigned char* mask = (const unsigned char*)d_in[2];
    const float* rd     = (const float*)d_in[3];
    const float* q_w    = (const float*)d_in[4];
    const float* q_b    = (const float*)d_in[5];
    const float* kv_w   = (const float*)d_in[6];
    const float* kv_b   = (const float*)d_in[7];
    const float* cpb_w1 = (const float*)d_in[8];
    const float* cpb_w2 = (const float*)d_in[9];
    const float* out_w  = (const float*)d_in[10];
    const float* out_b  = (const float*)d_in[11];
    float* out = (float*)d_out;

    char* p = (char*)d_ws;
    auto carve = [&](size_t halfs) {
        _Float16* r = (_Float16*)p;
        p += (halfs * sizeof(_Float16) + 255) & ~(size_t)255;
        return r;
    };
    _Float16* nodes16  = carve((size_t)Bn * Qn * Dn);       // 1,048,576
    _Float16* images16 = carve((size_t)Bn * Kn * Dn);       // 2,097,152
    _Float16* qw16     = carve((size_t)Dn * Dn);
    _Float16* kvw16    = carve((size_t)2 * Dn * Dn);
    _Float16* ow16     = carve((size_t)Dn * Dn);
    _Float16* w2pk     = carve(16 * 32 * 16);
    _Float16* qp       = carve((size_t)Bn * Qn * Dn);
    _Float16* kmat     = carve((size_t)Bn * Kn * Dn);
    _Float16* vT       = carve((size_t)Bn * Dn * Kn);
    _Float16* attn     = carve((size_t)Bn * Qn * Dn);

    auto cvt = [&](const float* src, _Float16* dst, int nelem) {
        cvt_f16_kernel<<<(nelem + 255) / 256, 256, 0, stream>>>(src, dst, nelem);
    };
    cvt(nodes,  nodes16,  Bn * Qn * Dn);
    cvt(images, images16, Bn * Kn * Dn);
    cvt(q_w,    qw16,     Dn * Dn);
    cvt(kv_w,   kvw16,    2 * Dn * Dn);
    cvt(out_w,  ow16,     Dn * Dn);
    pack_w2_kernel<<<(16 * 32 * 16 + 255) / 256, 256, 0, stream>>>(cpb_w2, w2pk);

    // (B*Q/16) * (D/16) = 4096 tiles, 8 waves/block
    qproj_kernel<<<512, 256, 0, stream>>>(nodes16, qw16, q_b, qp);
    // (B*K/16) * (2D/16) = 16384 tiles
    kvproj_kernel<<<2048, 256, 0, stream>>>(images16, kvw16, kv_b, kmat, vT);

    attn_kernel<<<dim3(Qn / 16, Bn), 256, 0, stream>>>(
        qp, kmat, vT, rd, cpb_w1, w2pk, mask, attn);

    oproj_kernel<<<512, 256, 0, stream>>>(attn, ow16, out_b, out);
}
